// PDTSPModel_unvisted_MLP_padavg_50079318671390
// MI455X (gfx1250) — matrix operand order repaired and checked by reading, not compiled, not run on hardware
//
#include <hip/hip_runtime.h>

// Problem constants (match reference)
#define B_   128
#define R_   500
#define N_   500
#define K_   10
#define E_   128
#define H1_  640
#define KE_  1280           // K_*E_ : MLP input width

// Tiling
#define ROWS   64           // rows of R per workgroup (four WMMA M tiles)
#define MT     4            // M tiles per workgroup
#define EMB_LD 1288         // f16 leading dim for emb tile (+8 pad -> 16B, bank-spread)
#define H_LD   648          // f16 leading dim for hidden tile (+8 pad)
#define EMB_ELEMS (ROWS * EMB_LD)               // 82,432 f16  (164,864 B)
#define H_ELEMS   (ROWS * H_LD)                 // 41,472 f16  ( 82,944 B)
#define SMEM_BYTES ((EMB_ELEMS + H_ELEMS) * 2)  // 247,808 B  (< 320 KB WGP budget)

typedef _Float16 v16h __attribute__((ext_vector_type(16)));
typedef _Float16 v8h  __attribute__((ext_vector_type(8)));
typedef _Float16 v4h  __attribute__((ext_vector_type(4)));
typedef float    v8f  __attribute__((ext_vector_type(8)));
typedef float    v4f  __attribute__((ext_vector_type(4)));

// ---------------------------------------------------------------------------
// Kernel 1: transpose + downconvert weights to f16 once per launch.
//   w1 (H1 x KE) f32  -> w1t (KE x H1) f16
//   w2 (E  x H1) f32  -> w2t (H1 x E ) f16
// Layout makes the 16 lanes of a half-wave read 16 contiguous f16 (32B) per B k row.
// ---------------------------------------------------------------------------
__global__ void PDTSP_convert_weights_kernel(const float* __restrict__ w1,
                                             const float* __restrict__ w2,
                                             _Float16* __restrict__ w1t,
                                             _Float16* __restrict__ w2t) {
    const int i  = blockIdx.x * blockDim.x + threadIdx.x;
    const int n1 = KE_ * H1_;
    if (i < n1) {
        const int k = i / H1_;
        const int h = i % H1_;
        w1t[i] = (_Float16)w1[h * KE_ + k];
    } else {
        const int o = i - n1;
        if (o < H1_ * E_) {
            const int k = o / E_;
            const int e = o % E_;
            w2t[o] = (_Float16)w2[e * H1_ + k];
        }
    }
}

// ---------------------------------------------------------------------------
// Kernel 2: fused top-K / gather / pad-avg / 2-layer MLP for a 64-row tile.
// 256 threads = 8 waves (wave32). Phase 1 builds the f16 emb tile in LDS in
// WMMA A-layout order; phases 2/3 run the GEMMs with v_wmma_f32_16x16x32_f16,
// reusing each B fragment across FOUR M tiles (4x arithmetic intensity on the
// L2-resident weights). 248 KB LDS -> relies on CDNA5's 320 KB/WGP budget.
// ---------------------------------------------------------------------------
__global__ __launch_bounds__(256)
void PDTSPModel_unvisted_MLP_padavg_50079318671390_kernel(
        const int*      __restrict__ current,
        const float*    __restrict__ distance,
        const float*    __restrict__ masked,
        const float*    __restrict__ encoded,
        const _Float16* __restrict__ w1t,
        const float*    __restrict__ b1,
        const _Float16* __restrict__ w2t,
        const float*    __restrict__ b2,
        float*          __restrict__ out)
{
    extern __shared__ _Float16 smem[];          // dynamic: 247,808 B per WG
    _Float16* embS = smem;                      // [ROWS][EMB_LD]
    _Float16* hS   = smem + EMB_ELEMS;          // [ROWS][H_LD] (16B-aligned offset)

    const int b    = blockIdx.y;
    const int r0   = blockIdx.x * ROWS;
    const int tid  = threadIdx.x;
    const int lane = tid & 31;
    const int wv   = tid >> 5;        // wave id 0..7

    // ---------------- Phase 1: top-K + gather + pad_avg --------------------
    // Each wave owns 8 rows; 500 candidates live in 16 regs/lane.
    for (int i = 0; i < 8; ++i) {
        const int rl = wv * 8 + i;
        int r = r0 + rl;
        if (r >= R_) r = R_ - 1;                 // clamp; duplicate rows not stored
        const int cur = current[b * R_ + r];
        const float* drow = distance + ((size_t)b * N_ + cur) * (size_t)N_;
        const float* mrow = masked   + ((size_t)b * R_ + r)   * (size_t)N_;

        float dv[16];
        #pragma unroll
        for (int j = 0; j < 16; ++j) {
            const int n = lane + 32 * j;
            float d = __builtin_inff();
            if (n < N_) {
                const float m = mrow[n];
                d = (m != -__builtin_inff()) ? drow[n] : __builtin_inff();
            }
            dv[j] = d;
        }

        int idxs[K_];
        #pragma unroll
        for (int t = 0; t < K_; ++t) {
            float best = __builtin_inff();
            int   bn   = N_;
            #pragma unroll
            for (int j = 0; j < 16; ++j) {
                const int n = lane + 32 * j;
                if (dv[j] < best || (dv[j] == best && n < bn)) { best = dv[j]; bn = n; }
            }
            #pragma unroll
            for (int off = 16; off > 0; off >>= 1) {   // wave32 argmin reduce
                const float ov = __shfl_xor(best, off, 32);
                const int   on = __shfl_xor(bn,   off, 32);
                if (ov < best || (ov == best && on < bn)) { best = ov; bn = on; }
            }
            const bool found = (best != __builtin_inff());
            idxs[t] = found ? bn : N_;
            if (found) {
                #pragma unroll
                for (int j = 0; j < 16; ++j)
                    if (lane + 32 * j == bn) dv[j] = __builtin_inff();
            }
        }

        // Gather: each lane covers 4 consecutive embedding elems (32*4 = 128).
        const int e0 = lane * 4;
        v4f sum  = {0.f, 0.f, 0.f, 0.f};
        v4f vals[K_];
        int nz = 0;
        #pragma unroll
        for (int s = 0; s < K_; ++s) {
            v4f val = {0.f, 0.f, 0.f, 0.f};
            const int id = idxs[s];
            if (id < N_) {
                val = *(const v4f*)(encoded + ((size_t)b * N_ + id) * E_ + e0);
                sum += val;
                ++nz;
            }
            vals[s] = val;
        }
        const float inv = (nz > 0) ? 1.0f / (float)nz : 0.0f;  // nz==0 -> zeros (matches clip(...,1e-9))
        v4f mean;
        #pragma unroll
        for (int q = 0; q < 4; ++q) mean[q] = sum[q] * inv;

        _Float16* erow = embS + rl * EMB_LD;
        #pragma unroll
        for (int s = 0; s < K_; ++s) {
            const v4f w = (idxs[s] < N_) ? vals[s] : mean;      // pad_avg
            v4h hv;
            #pragma unroll
            for (int q = 0; q < 4; ++q) hv[q] = (_Float16)w[q];
            *(v4h*)(erow + s * E_ + e0) = hv;                   // ds_store_b64
        }
    }
    __syncthreads();

    const int sel  = (lane < 16) ? 0 : 8;       // WMMA A K-half select
    const int rsel = (lane >> 4) << 3;          // WMMA C/D row offset (0 or 8)
    const int ln   = lane & 15;

    // ---------------- Phase 2: GEMM1  h = relu(emb x w1t + b1) -------------
    // Each wave: 4 M tiles x 5 N tiles; one B fragment feeds all four M tiles.
    {
        const _Float16* aBase[MT];
        #pragma unroll
        for (int m = 0; m < MT; ++m) aBase[m] = embS + (m * 16 + ln) * EMB_LD;
        const int nb = wv * 80;                  // 5 N-tiles of 16 per wave
        v8f acc[MT][5] = {};
        for (int kk = 0; kk < KE_; kk += 32) {
            v16h a[MT];
            #pragma unroll
            for (int m = 0; m < MT; ++m) {
                const v8h lo = *(const v8h*)(aBase[m] + kk + sel);       // ds_load_b128
                const v8h hi = *(const v8h*)(aBase[m] + kk + 16 + sel);
                #pragma unroll
                for (int q = 0; q < 8; ++q) { a[m][q] = lo[q]; a[m][q + 8] = hi[q]; }
            }
            const _Float16* bRow = w1t + (size_t)(kk + sel) * H1_ + nb + ln;
            __builtin_prefetch(bRow + 32 * H1_, 0, 1);                   // global_prefetch next K block
            #pragma unroll
            for (int j = 0; j < 5; ++j) {
                v16h bf;
                #pragma unroll
                for (int q = 0; q < 8; ++q) { bf[q] = bRow[q * H1_ + j * 16]; bf[q + 8] = bRow[(16 + q) * H1_ + j * 16]; }
                #pragma unroll
                for (int m = 0; m < MT; ++m) {
                    acc[m][j] = __builtin_amdgcn_wmma_f32_16x16x32_f16(
                                    false, a[m], false, bf, (short)0, acc[m][j], false, false);
                }
            }
        }
        #pragma unroll
        for (int j = 0; j < 5; ++j) {
            const int col  = nb + j * 16 + ln;
            const float bs = b1[col];
            #pragma unroll
            for (int m = 0; m < MT; ++m) {
                #pragma unroll
                for (int v = 0; v < 8; ++v) {
                    float x = acc[m][j][v] + bs;
                    x = x > 0.f ? x : 0.f;                               // relu
                    hS[(m * 16 + v + rsel) * H_LD + col] = (_Float16)x;
                }
            }
        }
    }
    __syncthreads();

    // ---------------- Phase 3: GEMM2  out = h x w2t + b2 -------------------
    // Each wave: one 16-col N tile, all four M tiles (B reused across M).
    {
        const _Float16* aBase[MT];
        #pragma unroll
        for (int m = 0; m < MT; ++m) aBase[m] = hS + (m * 16 + ln) * H_LD;
        const int nb = wv * 16;
        v8f acc[MT] = {};
        for (int kk = 0; kk < H1_; kk += 32) {
            v16h a[MT];
            #pragma unroll
            for (int m = 0; m < MT; ++m) {
                const v8h lo = *(const v8h*)(aBase[m] + kk + sel);
                const v8h hi = *(const v8h*)(aBase[m] + kk + 16 + sel);
                #pragma unroll
                for (int q = 0; q < 8; ++q) { a[m][q] = lo[q]; a[m][q + 8] = hi[q]; }
            }
            const _Float16* bCol = w2t + (size_t)(kk + sel) * E_ + nb + ln;
            v16h bf;
            #pragma unroll
            for (int q = 0; q < 8; ++q) { bf[q] = bCol[q * E_]; bf[q + 8] = bCol[(16 + q) * E_]; }
            #pragma unroll
            for (int m = 0; m < MT; ++m) {
                acc[m] = __builtin_amdgcn_wmma_f32_16x16x32_f16(
                             false, a[m], false, bf, (short)0, acc[m], false, false);
            }
        }
        const int col  = nb + ln;
        const float bs = b2[col];
        #pragma unroll
        for (int m = 0; m < MT; ++m) {
            #pragma unroll
            for (int v = 0; v < 8; ++v) {
                const int r = r0 + m * 16 + v + rsel;
                if (r < R_)
                    out[((size_t)b * R_ + r) * E_ + col] = acc[m][v] + bs;
            }
        }
    }
}

extern "C" void kernel_launch(void* const* d_in, const int* in_sizes, int n_in,
                              void* d_out, int out_size, void* d_ws, size_t ws_size,
                              hipStream_t stream) {
    const int*   current  = (const int*)  d_in[0];
    const float* distance = (const float*)d_in[1];
    const float* masked   = (const float*)d_in[2];
    const float* encoded  = (const float*)d_in[3];
    const float* w1       = (const float*)d_in[4];
    const float* b1       = (const float*)d_in[5];
    const float* w2       = (const float*)d_in[6];
    const float* b2       = (const float*)d_in[7];
    float* out = (float*)d_out;

    // Workspace: f16 transposed weights (1,638,400 B + 163,840 B ~= 1.8 MB)
    _Float16* w1t = (_Float16*)d_ws;
    _Float16* w2t = w1t + (size_t)KE_ * H1_;

    const int totalW = KE_ * H1_ + H1_ * E_;
    PDTSP_convert_weights_kernel<<<(totalW + 255) / 256, 256, 0, stream>>>(w1, w2, w1t, w2t);

    dim3 grid((R_ + ROWS - 1) / ROWS, B_);     // 8 x 128 workgroups
    PDTSPModel_unvisted_MLP_padavg_50079318671390_kernel<<<grid, 256, SMEM_BYTES, stream>>>(
        current, distance, masked, encoded, w1t, b1, w2t, b2, out);
}
